// AttentionHead_1967095022255
// MI455X (gfx1250) — matrix operand order repaired and checked by reading, not compiled, not run on hardware
//
#include <hip/hip_runtime.h>
#include <hip/hip_bf16.h>

typedef _Float16 half_t;
typedef __attribute__((ext_vector_type(16))) _Float16 v16h;
typedef __attribute__((ext_vector_type(8)))  float    v8f;

#define D_EMBED 1024
#define HEAD    64
#define BATCH   8
#define SEQ     2048

// 32-byte fragment: 16 f16 values per lane (A or B operand of 16x16x32 WMMA)
union HFrag {
    v16h   v;
    half_t h[16];
    uint4  q[2];
};

__device__ __forceinline__ v8f wmma_f16(const HFrag& a, const HFrag& b, v8f c) {
    // D = A(16x32 f16) * B(32x16 f16) + C(16x16 f32)
    return __builtin_amdgcn_wmma_f32_16x16x32_f16(
        /*neg_a=*/false, a.v, /*neg_b=*/false, b.v,
        /*c_mod=*/(short)0, c, /*reuse_a=*/false, /*reuse_b=*/false);
}

// ---------------------------------------------------------------------------
// Pass 0: W [1024,64] f32  ->  W^T [64,1024] f16   (x3 matrices)
// ---------------------------------------------------------------------------
__global__ void wt_convert_kernel(const float* __restrict__ Wq,
                                  const float* __restrict__ Wk,
                                  const float* __restrict__ Wv,
                                  half_t* __restrict__ WqT,
                                  half_t* __restrict__ WkT,
                                  half_t* __restrict__ WvT) {
    int idx = blockIdx.x * blockDim.x + threadIdx.x;   // 0 .. 3*65536-1
    int m = idx >> 16;
    int r = idx & 65535;
    int h = r >> 10;        // 0..63
    int d = r & 1023;       // 0..1023
    const float* W  = (m == 0) ? Wq  : (m == 1) ? Wk  : Wv;
    half_t*      WT = (m == 0) ? WqT : (m == 1) ? WkT : WvT;
    WT[h * D_EMBED + d] = (half_t)W[d * HEAD + h];
}

// ---------------------------------------------------------------------------
// Pass 1: Q,K = x*Wq, x*Wk (f16, [B*S,64]); V^T = (x*Wv)^T (f16, [B,64,S])
// One wave per 16-row tile; 8 waves per block.
// dc loop unrolled 2x with double-buffered A *and* B fragments: phase-P
// WMMAs consume data whose loads were issued one phase earlier, so loads
// stay in flight across the WMMA groups (no full loadcnt drains).
// ---------------------------------------------------------------------------
__global__ void __launch_bounds__(256)
qkv_kernel(const float* __restrict__ x,
           const half_t* __restrict__ WqT,
           const half_t* __restrict__ WkT,
           const half_t* __restrict__ WvT,
           half_t* __restrict__ Q,
           half_t* __restrict__ K,
           half_t* __restrict__ VT) {
    const int lane = threadIdx.x & 31;
    const int wave = threadIdx.x >> 5;
    const int row0 = (blockIdx.x * 8 + wave) * 16;
    const int l16  = lane & 15;
    const int hi   = lane >> 4;

    v8f zero = {};
    v8f accQ[4], accK[4], accV[4];
#pragma unroll
    for (int t = 0; t < 4; ++t) { accQ[t] = zero; accK[t] = zero; accV[t] = zero; }

    // A source: row = row0 + l16, runs at K = hi*8 and 16 + hi*8
    const float* xr = x + (size_t)(row0 + l16) * D_EMBED + hi * 8;
    // B source row offset (lane = output col l16, runs at K = hi*8 and 16 + hi*8)
    const size_t wrow = (size_t)l16 * D_EMBED + hi * 8;

    auto loadX = [&](int dc, float4* f) {
        const float* p = xr + dc;
        f[0] = *reinterpret_cast<const float4*>(p);
        f[1] = *reinterpret_cast<const float4*>(p + 4);
        f[2] = *reinterpret_cast<const float4*>(p + 16);
        f[3] = *reinterpret_cast<const float4*>(p + 20);
    };
    auto loadB = [&](const half_t* WT, int dc, HFrag* b) {
#pragma unroll
        for (int ht = 0; ht < 4; ++ht) {
            const half_t* p = WT + wrow + (size_t)ht * 16 * D_EMBED + dc;
            b[ht].q[0] = *reinterpret_cast<const uint4*>(p);
            b[ht].q[1] = *reinterpret_cast<const uint4*>(p + 16);
        }
    };
    auto cvtA = [&](const float4* f, HFrag& a) {
        a.h[0]  = (half_t)f[0].x; a.h[1]  = (half_t)f[0].y; a.h[2]  = (half_t)f[0].z; a.h[3]  = (half_t)f[0].w;
        a.h[4]  = (half_t)f[1].x; a.h[5]  = (half_t)f[1].y; a.h[6]  = (half_t)f[1].z; a.h[7]  = (half_t)f[1].w;
        a.h[8]  = (half_t)f[2].x; a.h[9]  = (half_t)f[2].y; a.h[10] = (half_t)f[2].z; a.h[11] = (half_t)f[2].w;
        a.h[12] = (half_t)f[3].x; a.h[13] = (half_t)f[3].y; a.h[14] = (half_t)f[3].z; a.h[15] = (half_t)f[3].w;
    };

    float4 fA[4], fB[4];
    HFrag bqA[4], bkA[4], bvA[4];
    HFrag bqB[4], bkB[4], bvB[4];

    // pipeline prologue: phase-A data for dc = 0
    loadX(0, fA);
    loadB(WqT, 0, bqA); loadB(WkT, 0, bkA); loadB(WvT, 0, bvA);

    for (int dc = 0; dc < D_EMBED; dc += 64) {
        // issue phase-B loads (dc+32 <= 992 always valid) before phase-A math
        loadX(dc + 32, fB);
        loadB(WqT, dc + 32, bqB); loadB(WkT, dc + 32, bkB); loadB(WvT, dc + 32, bvB);
        __builtin_prefetch(xr + dc + 64, 0, 0);

        HFrag aA; cvtA(fA, aA);
#pragma unroll
        for (int ht = 0; ht < 4; ++ht) {
            accQ[ht] = wmma_f16(aA, bqA[ht], accQ[ht]);
            accK[ht] = wmma_f16(aA, bkA[ht], accK[ht]);
            accV[ht] = wmma_f16(aA, bvA[ht], accV[ht]);
        }

        // issue next-iteration phase-A loads before phase-B math
        if (dc + 64 < D_EMBED) {
            loadX(dc + 64, fA);
            loadB(WqT, dc + 64, bqA); loadB(WkT, dc + 64, bkA); loadB(WvT, dc + 64, bvA);
        }

        HFrag aB; cvtA(fB, aB);
#pragma unroll
        for (int ht = 0; ht < 4; ++ht) {
            accQ[ht] = wmma_f16(aB, bqB[ht], accQ[ht]);
            accK[ht] = wmma_f16(aB, bkB[ht], accK[ht]);
            accV[ht] = wmma_f16(aB, bvB[ht], accV[ht]);
        }
    }

    // ---- epilogue: C layout row = r + 8*hi, col = ht*16 + l16
#pragma unroll
    for (int ht = 0; ht < 4; ++ht) {
        const int h = ht * 16 + l16;
#pragma unroll
        for (int r = 0; r < 8; ++r) {
            const int g = row0 + r + 8 * hi;          // flat row in [0, B*S)
            Q[(size_t)g * HEAD + h] = (half_t)accQ[ht][r];
            K[(size_t)g * HEAD + h] = (half_t)accK[ht][r];
            const int bb = g >> 11, ss = g & (SEQ - 1);
            VT[(((size_t)bb * HEAD + h) << 11) + ss] = (half_t)accV[ht][r];
        }
    }
}

// ---------------------------------------------------------------------------
// Pass 2: fused causal flash attention. One wave per 16-query tile.
// Pipeline order per iteration: score WMMAs (K loaded last iteration) ->
// V loads -> next K loads -> softmax VALU (covers load latency) -> PV WMMAs.
// ---------------------------------------------------------------------------
#define LDS_STRIDE 40   // halves per row (32 + pad)

__global__ void __launch_bounds__(128)
attn_kernel(const half_t* __restrict__ Q,
            const half_t* __restrict__ K,
            const half_t* __restrict__ VT,
            float* __restrict__ out) {
    __shared__ half_t lds[4 * 16 * LDS_STRIDE];

    const int lane  = threadIdx.x & 31;
    const int wave  = threadIdx.x >> 5;
    const int l16   = lane & 15;
    const int hi    = lane >> 4;
    const int qtile = blockIdx.x * 4 + wave;
    const int b     = qtile >> 7;            // / (SEQ/16)
    const int q0    = (qtile & 127) * 16;

    const half_t* Qb = Q  + (size_t)b * SEQ * HEAD;
    const half_t* Kb = K  + (size_t)b * SEQ * HEAD;
    const half_t* Vb = VT + (size_t)b * HEAD * SEQ + (size_t)l16 * SEQ + hi * 16;
    half_t* myLds = lds + wave * 16 * LDS_STRIDE;

    // Q A-fragments for the two 32-wide chunks of H
    HFrag aQ0, aQ1;
    {
        const half_t* qr = Qb + (size_t)(q0 + l16) * HEAD + hi * 16;
        aQ0.q[0] = *reinterpret_cast<const uint4*>(qr);
        aQ0.q[1] = *reinterpret_cast<const uint4*>(qr + 8);
        aQ1.q[0] = *reinterpret_cast<const uint4*>(qr + 32);
        aQ1.q[1] = *reinterpret_cast<const uint4*>(qr + 40);
    }

    HFrag bk[4];
    auto loadK = [&](int kb) {
        const half_t* kr0 = Kb + (size_t)(kb + l16) * HEAD + hi * 16;
        const half_t* kr1 = kr0 + 16 * HEAD;
        bk[0].q[0] = *reinterpret_cast<const uint4*>(kr0);
        bk[0].q[1] = *reinterpret_cast<const uint4*>(kr0 + 8);
        bk[1].q[0] = *reinterpret_cast<const uint4*>(kr0 + 32);
        bk[1].q[1] = *reinterpret_cast<const uint4*>(kr0 + 40);
        bk[2].q[0] = *reinterpret_cast<const uint4*>(kr1);
        bk[2].q[1] = *reinterpret_cast<const uint4*>(kr1 + 8);
        bk[3].q[0] = *reinterpret_cast<const uint4*>(kr1 + 32);
        bk[3].q[1] = *reinterpret_cast<const uint4*>(kr1 + 40);
    };

    v8f zero = {};
    v8f accO[4];
    float m_i[8], l_i[8];
#pragma unroll
    for (int t = 0; t < 4; ++t) accO[t] = zero;
#pragma unroll
    for (int r = 0; r < 8; ++r) { m_i[r] = -1e30f; l_i[r] = 0.0f; }

    const float scale = 0.125f;   // 1/sqrt(64)
    const int kend = q0 + 16;     // causal limit (exclusive)

    loadK(0);                     // pipeline prologue

    for (int kb = 0; kb < kend; kb += 32) {
        // ---- scores: two 16x16 tiles, inner dim H=64 (2 WMMAs each)
        v8f s0 = zero, s1 = zero;
        s0 = wmma_f16(aQ0, bk[0], s0);
        s0 = wmma_f16(aQ1, bk[1], s0);
        s1 = wmma_f16(aQ0, bk[2], s1);
        s1 = wmma_f16(aQ1, bk[3], s1);

        // ---- issue V loads for this block (independent of softmax result)
        HFrag bv[4];
#pragma unroll
        for (int ht = 0; ht < 4; ++ht) {
            const half_t* vr = Vb + (size_t)ht * 16 * SEQ + kb;
            bv[ht].q[0] = *reinterpret_cast<const uint4*>(vr);
            bv[ht].q[1] = *reinterpret_cast<const uint4*>(vr + 8);
        }

        // ---- issue next-iteration K loads (land during softmax + PV)
        if (kb + 32 < kend) loadK(kb + 32);

        // ---- online softmax over rows r+8*hi (row = 16 lanes of one half)
        float corr[8];
#pragma unroll
        for (int r = 0; r < 8; ++r) {
            const int rowM = q0 + r + 8 * hi;
            float e0 = s0[r] * scale; if (kb + l16      > rowM) e0 = -1e30f;
            float e1 = s1[r] * scale; if (kb + 16 + l16 > rowM) e1 = -1e30f;
            float rm = fmaxf(e0, e1);
            rm = fmaxf(rm, __shfl_xor(rm, 1, 32));
            rm = fmaxf(rm, __shfl_xor(rm, 2, 32));
            rm = fmaxf(rm, __shfl_xor(rm, 4, 32));
            rm = fmaxf(rm, __shfl_xor(rm, 8, 32));
            float mn = fmaxf(m_i[r], rm);
            float p0 = __expf(e0 - mn);
            float p1 = __expf(e1 - mn);
            float ps = p0 + p1;
            ps += __shfl_xor(ps, 1, 32);
            ps += __shfl_xor(ps, 2, 32);
            ps += __shfl_xor(ps, 4, 32);
            ps += __shfl_xor(ps, 8, 32);
            float c = __expf(m_i[r] - mn);
            l_i[r] = l_i[r] * c + ps;
            m_i[r] = mn;
            corr[r] = c;
            // write P (f16) to LDS at [row][col] for the A-layout transpose
            myLds[(r + 8 * hi) * LDS_STRIDE + l16]      = (half_t)p0;
            myLds[(r + 8 * hi) * LDS_STRIDE + 16 + l16] = (half_t)p1;
        }
#pragma unroll
        for (int t = 0; t < 4; ++t)
#pragma unroll
            for (int r = 0; r < 8; ++r) accO[t][r] = accO[t][r] * corr[r];

        // wave-private LDS transpose: wait for ds stores, read back as A frag
        asm volatile("s_wait_dscnt 0" ::: "memory");
        HFrag aP;
        {
            const half_t* pr = myLds + l16 * LDS_STRIDE + hi * 16;
            aP.q[0] = *reinterpret_cast<const uint4*>(pr);
            aP.q[1] = *reinterpret_cast<const uint4*>(pr + 8);
        }

        // ---- O += P * V  (V fragments long since in flight)
#pragma unroll
        for (int ht = 0; ht < 4; ++ht) {
            accO[ht] = wmma_f16(aP, bv[ht], accO[ht]);
        }
    }

    // ---- epilogue: out = O * (1/l), f32 [B,S,H]
#pragma unroll
    for (int r = 0; r < 8; ++r) l_i[r] = __builtin_amdgcn_rcpf(l_i[r]);

    float* ob = out + (size_t)b * SEQ * HEAD;
#pragma unroll
    for (int ht = 0; ht < 4; ++ht) {
        const int h = ht * 16 + l16;
#pragma unroll
        for (int r = 0; r < 8; ++r) {
            const int s = q0 + r + 8 * hi;
            ob[(size_t)s * HEAD + h] = accO[ht][r] * l_i[r];
        }
    }
}

// ---------------------------------------------------------------------------
extern "C" void kernel_launch(void* const* d_in, const int* in_sizes, int n_in,
                              void* d_out, int out_size, void* d_ws, size_t ws_size,
                              hipStream_t stream) {
    (void)in_sizes; (void)n_in; (void)out_size; (void)ws_size;
    const float* x  = (const float*)d_in[0];
    const float* Wq = (const float*)d_in[1];
    const float* Wk = (const float*)d_in[2];
    const float* Wv = (const float*)d_in[3];

    char* ws = (char*)d_ws;
    const size_t WT_BYTES = (size_t)HEAD * D_EMBED * sizeof(half_t);     // 128 KB
    const size_t QK_BYTES = (size_t)BATCH * SEQ * HEAD * sizeof(half_t); // 2 MB
    half_t* WqT = (half_t*)(ws);
    half_t* WkT = (half_t*)(ws + WT_BYTES);
    half_t* WvT = (half_t*)(ws + 2 * WT_BYTES);
    half_t* Qh  = (half_t*)(ws + 3 * WT_BYTES);
    half_t* Kh  = (half_t*)(ws + 3 * WT_BYTES + QK_BYTES);
    half_t* VTh = (half_t*)(ws + 3 * WT_BYTES + 2 * QK_BYTES);

    wt_convert_kernel<<<768, 256, 0, stream>>>(Wq, Wk, Wv, WqT, WkT, WvT);
    qkv_kernel<<<(BATCH * SEQ) / 128, 256, 0, stream>>>(x, WqT, WkT, WvT, Qh, Kh, VTh);
    attn_kernel<<<(BATCH * SEQ / 16) / 4, 128, 0, stream>>>(Qh, Kh, VTh, (float*)d_out);
}
